// Decoder_23235773071576
// MI455X (gfx1250) — matrix operand order repaired and checked by reading, not compiled, not run on hardware
//
#include <hip/hip_runtime.h>
#include <hip/hip_bf16.h>
#include <math.h>

typedef __attribute__((ext_vector_type(16))) __bf16 bf16x16;
typedef __attribute__((ext_vector_type(8)))  float   f32x8;

#define B_  256
#define T_  64
#define H_  256
#define M2_ 512

// One 16x16 C tile of  C = A(16xK, f32 row-major, stride lda) * W^T,
// W pre-converted to bf16 (row-major, stride ldw): B element (k,n) = W[n*ldw+k].
// Fragment layouts per CDNA5 ISA 7.12.2 (16-bit A 16x32, B 32x16, f32 C 16x16):
//   A: lane half h, row r=lane&15; VGPR v holds K = (v<4 ? 2v : 16+2(v-4)) + 8h (+e)
//   B: lane col  n=lane&15;        VGPR v holds K = (v<4 ? 2v : 16+2(v-4)) + 8h (+e)
__device__ __forceinline__ f32x8 wmma_f32a_bf16w(const float* A, int lda,
                                                 const __bf16* W, int ldw,
                                                 int K, int lane, f32x8 acc)
{
  const int half = lane >> 4;
  const int r    = lane & 15;
  for (int k0 = 0; k0 < K; k0 += 32) {
    bf16x16 af, bf;
#pragma unroll
    for (int v = 0; v < 8; ++v) {
      const int ka = (v < 4 ? 2*v : 16 + 2*(v-4)) + half*8 + k0;
      af[2*v]   = (__bf16)A[r*lda + ka];
      af[2*v+1] = (__bf16)A[r*lda + ka + 1];
      const int kb = (v < 4 ? 2*v : 16 + 2*(v-4)) + half*8 + k0;
      bf[2*v]   = W[r*ldw + kb];
      bf[2*v+1] = W[r*ldw + kb + 1];
    }
    acc = __builtin_amdgcn_wmma_f32_16x16x32_bf16(false, af, false, bf,
                                                  (short)0, acc, false, false);
  }
  return acc;
}

// ---------------------------------------------------------------------------
// Kernel 0: one-time f32 -> bf16 weight pack (removes per-step cvt + halves
// the recurrent L2 weight stream).
// ---------------------------------------------------------------------------
__global__ void __launch_bounds__(256)
cvt_bf16_kernel(const float* __restrict__ src, __bf16* __restrict__ dst, int n)
{
  const int i = blockIdx.x * 256 + threadIdx.x;
  if (i < n) dst[i] = (__bf16)src[i];
}

// ---------------------------------------------------------------------------
// Kernel 1: E2[bt,n] = (IE[bt,:] @ W1e^T) @ W2^T  (step-invariant attention
// term; 16.8 MB, L2-resident for the whole scan).
// ---------------------------------------------------------------------------
__global__ void __launch_bounds__(256)
e2_precompute(const float*  __restrict__ IE,
              const __bf16* __restrict__ w1b,   // (256,1024) bf16; cols [512,1024)=W1e
              const __bf16* __restrict__ w2b,   // (256,256) bf16
              float* __restrict__ E2)           // (B*T, 256)
{
  __shared__ float sE1[16 * 256];
  const int m0   = blockIdx.x * 16;
  const int wave = threadIdx.x >> 5;
  const int lane = threadIdx.x & 31;
  const int half = lane >> 4, nn = lane & 15;

  for (int nt = wave; nt < 16; nt += 8) {
    f32x8 acc = {};
    acc = wmma_f32a_bf16w(IE + (size_t)m0 * M2_, M2_,
                          w1b + (size_t)(nt * 16) * 1024 + M2_, 1024, M2_, lane, acc);
#pragma unroll
    for (int v = 0; v < 8; ++v)
      sE1[(v + half*8) * 256 + nt*16 + nn] = acc[v];
  }
  __syncthreads();
  for (int nt = wave; nt < 16; nt += 8) {
    f32x8 acc = {};
    acc = wmma_f32a_bf16w(sE1, 256, w2b + (size_t)(nt * 16) * 256, 256, 256, lane, acc);
#pragma unroll
    for (int v = 0; v < 8; ++v)
      E2[(size_t)(m0 + v + half*8) * 256 + nt*16 + nn] = acc[v];
  }
}

// ---------------------------------------------------------------------------
// Kernel 2: persistent scan. 16 WGs x 256 threads; WG g owns batch rows
// [16g,16g+16). All recurrent state in LDS; zero global sync across steps.
// ---------------------------------------------------------------------------
__global__ void __launch_bounds__(256)
decoder_scan(const float*  __restrict__ IE,   const float* __restrict__ yhist,
             const float*  __restrict__ spd,  const float* __restrict__ h0,
             const float*  __restrict__ c0,
             const __bf16* __restrict__ w1b,  const float* __restrict__ b1,
             const __bf16* __restrict__ w2b,  const float* __restrict__ w2,
             const float*  __restrict__ b2,
             const float*  __restrict__ w3,   const float* __restrict__ b3,
             const float*  __restrict__ wihf, const __bf16* __restrict__ whhfb,
             const float*  __restrict__ bihf, const float* __restrict__ bhhf,
             const float*  __restrict__ wihb, const __bf16* __restrict__ whhbb,
             const float*  __restrict__ bihb, const float* __restrict__ bhhb,
             const __bf16* __restrict__ fcw1b, const float* __restrict__ fcw1,
             const float*  __restrict__ fcb1,
             const float*  __restrict__ fcw2, const float* __restrict__ fcb2,
             const float*  __restrict__ fwin, const float* __restrict__ fbin,
             const float*  __restrict__ E2,
             float* __restrict__ out_w, float* __restrict__ out_y)
{
  __shared__ float sHC[2 * 16 * 512];   // [p][b][0:256]=h, [256:512]=c
  __shared__ float R[16 * 1024];        // reused: u/v -> ctx/tmp1 -> gates
  __shared__ float sScore[16 * 64];
  __shared__ float sAcc[16 * 64];
  __shared__ float sVB[256];            // b1 @ W2^T + b2   (step-invariant)
  __shared__ float sGB[2 * 1024];       // b_ih + b_hh per direction
  __shared__ float sYt[16];
  __shared__ float sOutPart[16];

  const int g    = blockIdx.x;
  const int tid  = threadIdx.x;
  const int wave = tid >> 5, lane = tid & 31;
  const int half = lane >> 4, nn = lane & 15;
  const int brow0 = g * 16;

  // ---- init: state, accumulators, folded bias vectors
  for (int i = tid; i < 2 * 16 * 256; i += 256) {
    const int p = i >> 12, b = (i >> 8) & 15, k = i & 255;
    sHC[(p*16 + b)*512 + k]       = h0[((size_t)p*B_ + brow0 + b)*H_ + k];
    sHC[(p*16 + b)*512 + 256 + k] = c0[((size_t)p*B_ + brow0 + b)*H_ + k];
  }
  for (int i = tid; i < 16 * 64; i += 256) sAcc[i] = 0.f;
  {   // sVB[n] = sum_k b1[k]*w2[n,k] + b2[n]
    float acc = 0.f;
    for (int k = 0; k < 256; ++k) acc += b1[k] * w2[tid * 256 + k];
    sVB[tid] = acc + b2[tid];
  }
  for (int i = tid; i < 2 * 1024; i += 256) {
    const int p = i >> 10, j = i & 1023;
    sGB[i] = (p ? bihb[j] + bhhb[j] : bihf[j] + bhhf[j]);
  }
  __syncthreads();

  for (int s = 0; s < T_; ++s) {
    // ---- A: u[p] = [h_p|c_p] @ W1hc^T   -> R[p*4096 + b*256 + n]
    for (int tile = wave; tile < 32; tile += 8) {
      const int p = tile >> 4, nt = tile & 15;
      f32x8 acc = {};
      acc = wmma_f32a_bf16w(&sHC[p*16*512], 512,
                            w1b + (size_t)(nt*16) * 1024, 1024, 512, lane, acc);
#pragma unroll
      for (int v = 0; v < 8; ++v)
        R[p*4096 + (v + half*8)*256 + nt*16 + nn] = acc[v];
    }
    __syncthreads();
    // ---- B: v[p] = u[p] @ W2^T + (b1@W2^T + b2)   -> R[8192 + ...]
    for (int tile = wave; tile < 32; tile += 8) {
      const int p = tile >> 4, nt = tile & 15;
      f32x8 acc = {};
      acc = wmma_f32a_bf16w(R + p*4096, 256,
                            w2b + (size_t)(nt*16) * 256, 256, 256, lane, acc);
#pragma unroll
      for (int v = 0; v < 8; ++v)
        R[8192 + p*4096 + (v + half*8)*256 + nt*16 + nn] = acc[v] + sVB[nt*16 + nn];
    }
    __syncthreads();
    // ---- C: scores[b,t] = sum_k tanh(v[t%2]+E2) * w3 + b3
    for (int pair = tid; pair < 16 * 64; pair += 256) {
      const int b = pair >> 6, t = pair & 63;
      const float* vrow  = R + 8192 + (t & 1)*4096 + b*256;
      const float* e2row = E2 + ((size_t)(brow0 + b)*T_ + t) * 256;
      float sum = 0.f;
      for (int k = 0; k < 256; ++k)
        sum += tanhf(vrow[k] + e2row[k]) * w3[k];
      sScore[b*64 + t] = sum + b3[0];
    }
    __syncthreads();
    // ---- D: softmax over t, accumulate attention weights
    if (tid < 16) {
      const int b = tid;
      float mx = -3.0e38f;
      for (int t = 0; t < 64; ++t) mx = fmaxf(mx, sScore[b*64 + t]);
      float sum = 0.f;
      for (int t = 0; t < 64; ++t) { float e = expf(sScore[b*64 + t] - mx); sScore[b*64 + t] = e; sum += e; }
      const float inv = 1.f / sum;
      for (int t = 0; t < 64; ++t) { float wv = sScore[b*64 + t] * inv; sScore[b*64 + t] = wv; sAcc[b*64 + t] += wv; }
    }
    __syncthreads();
    // ---- E: context[b,m] = attw @ IE   -> R[b*512 + m] (u/v now dead)
    for (int i = tid; i < 16 * 512; i += 256) {
      const int b = i >> 9, m = i & 511;
      const float* ie = IE + (size_t)(brow0 + b) * T_ * M2_ + m;
      float sum = 0.f;
      for (int t = 0; t < 64; ++t) sum += sScore[b*64 + t] * ie[(size_t)t * M2_];
      R[b*512 + m] = sum;
    }
    __syncthreads();
    // ---- F: tmp1 = [ctx|y|s] @ fcw1^T  (WMMA K=512 + 2-col remainder + bias)
    for (int nt = wave; nt < 16; nt += 8) {
      f32x8 acc = {};
      acc = wmma_f32a_bf16w(R, 512, fcw1b + (size_t)(nt*16) * 514, 514, 512, lane, acc);
#pragma unroll
      for (int v = 0; v < 8; ++v) {
        const int b = v + half*8, n = nt*16 + nn;
        const float yb = yhist[(size_t)(brow0 + b)*T_ + s];
        const float sb = spd  [(size_t)(brow0 + b)*T_ + s];
        R[8192 + b*256 + n] = acc[v] + yb*fcw1[n*514 + 512] + sb*fcw1[n*514 + 513] + fcb1[n];
      }
    }
    __syncthreads();
    if (tid < 16) {
      const int b = tid;
      float yt = 0.f;
      for (int n = 0; n < 256; ++n) yt += R[8192 + b*256 + n] * fcw2[n];
      sYt[b] = yt + fcb2[0];
      if (s == T_ - 1) {                       // save final-context partial of out
        float op = 0.f;
        for (int m = 0; m < 512; ++m) op += R[b*512 + m] * fwin[256 + m];
        sOutPart[b] = op;
      }
    }
    __syncthreads();
    // ---- G: LSTM cells (p=0 fwd, p=1 bwd): gates = h_p @ whh^T + yt*wih + b
    for (int p = 0; p < 2; ++p) {
      const __bf16* whh = p ? whhbb : whhfb;
      const float*  wih = p ? wihb  : wihf;
      for (int nt = wave; nt < 64; nt += 8) {
        f32x8 acc = {};
        acc = wmma_f32a_bf16w(&sHC[p*16*512], 512,
                              whh + (size_t)(nt*16) * 256, 256, 256, lane, acc);
#pragma unroll
        for (int v = 0; v < 8; ++v) {
          const int b = v + half*8, j = nt*16 + nn;
          R[b*1024 + j] = acc[v] + sYt[b]*wih[j] + sGB[p*1024 + j];
        }
      }
      __syncthreads();
      for (int i = tid; i < 16 * 256; i += 256) {
        const int b = i >> 8, k = i & 255;
        const float zi = R[b*1024 + k];
        const float zf = R[b*1024 + 256 + k];
        const float zg = R[b*1024 + 512 + k];
        const float zo = R[b*1024 + 768 + k];
        const float c_old = sHC[(p*16 + b)*512 + 256 + k];
        const float si = 1.f / (1.f + expf(-zi));
        const float sf = 1.f / (1.f + expf(-zf));
        const float so = 1.f / (1.f + expf(-zo));
        const float cn = sf*c_old + si*tanhf(zg);
        sHC[(p*16 + b)*512 + 256 + k] = cn;
        sHC[(p*16 + b)*512 + k]       = so*tanhf(cn);
      }
      __syncthreads();
    }
  }

  // ---- finalize: weights = sum(attw)/T ; out = [h_f | ctx_last] @ fcfin^T + b
  for (int i = tid; i < 16 * 64; i += 256)
    out_w[(size_t)(brow0 + (i >> 6))*T_ + (i & 63)] = sAcc[i] * (1.0f / T_);
  if (tid < 16) {
    const int b = tid;
    float val = 0.f;
    for (int k = 0; k < 256; ++k) val += sHC[b*512 + k] * fwin[k];   // p=0 hidden
    out_y[brow0 + b] = val + sOutPart[b] + fbin[0];
  }
}

extern "C" void kernel_launch(void* const* d_in, const int* in_sizes, int n_in,
                              void* d_out, int out_size, void* d_ws, size_t ws_size,
                              hipStream_t stream)
{
  (void)in_sizes; (void)n_in; (void)out_size; (void)ws_size;

  const float* IE   = (const float*)d_in[0];
  const float* yh   = (const float*)d_in[1];
  const float* sp   = (const float*)d_in[2];
  const float* h0   = (const float*)d_in[3];
  const float* c0   = (const float*)d_in[4];
  const float* w1   = (const float*)d_in[5];
  const float* b1   = (const float*)d_in[6];
  const float* w2   = (const float*)d_in[7];
  const float* b2   = (const float*)d_in[8];
  const float* w3   = (const float*)d_in[9];
  const float* b3   = (const float*)d_in[10];
  const float* wihf = (const float*)d_in[11];
  const float* whhf = (const float*)d_in[12];
  const float* bihf = (const float*)d_in[13];
  const float* bhhf = (const float*)d_in[14];
  const float* wihb = (const float*)d_in[15];
  const float* whhb = (const float*)d_in[16];
  const float* bihb = (const float*)d_in[17];
  const float* bhhb = (const float*)d_in[18];
  const float* fcw1 = (const float*)d_in[19];
  const float* fcb1 = (const float*)d_in[20];
  const float* fcw2 = (const float*)d_in[21];
  const float* fcb2 = (const float*)d_in[22];
  const float* fwin = (const float*)d_in[23];
  const float* fbin = (const float*)d_in[24];

  // workspace layout
  char*   wsb   = (char*)d_ws;
  float*  E2    = (float*)wsb;                          // 16,777,216 B
  __bf16* w1b   = (__bf16*)(wsb + 16777216);            // 256*1024
  __bf16* w2b   = w1b   + 262144;                       // 256*256
  __bf16* fcw1b = w2b   + 65536;                        // 256*514
  __bf16* whhfb = fcw1b + 131584;                       // 1024*256
  __bf16* whhbb = whhfb + 262144;                       // 1024*256

  float* out_w = (float*)d_out;                         // (B, T)
  float* out_y = (float*)d_out + (size_t)B_ * T_;       // (B, 1)

  // one-time bf16 weight packs
  hipLaunchKernelGGL(cvt_bf16_kernel, dim3((262144 + 255)/256), dim3(256), 0, stream, w1,   w1b,   262144);
  hipLaunchKernelGGL(cvt_bf16_kernel, dim3(( 65536 + 255)/256), dim3(256), 0, stream, w2,   w2b,    65536);
  hipLaunchKernelGGL(cvt_bf16_kernel, dim3((131584 + 255)/256), dim3(256), 0, stream, fcw1, fcw1b, 131584);
  hipLaunchKernelGGL(cvt_bf16_kernel, dim3((262144 + 255)/256), dim3(256), 0, stream, whhf, whhfb, 262144);
  hipLaunchKernelGGL(cvt_bf16_kernel, dim3((262144 + 255)/256), dim3(256), 0, stream, whhb, whhbb, 262144);

  hipLaunchKernelGGL(e2_precompute, dim3((B_ * T_) / 16), dim3(256), 0, stream,
                     IE, w1b, w2b, E2);
  hipLaunchKernelGGL(decoder_scan, dim3(16), dim3(256), 0, stream,
                     IE, yh, sp, h0, c0,
                     w1b, b1, w2b, w2, b2, w3, b3,
                     wihf, whhfb, bihf, bhhf,
                     wihb, whhbb, bihb, bhhb,
                     fcw1b, fcw1, fcb1, fcw2, fcb2, fwin, fbin,
                     E2, out_w, out_y);
}
